// DyEdgeGAT_41240275976721
// MI455X (gfx1250) — matrix operand order repaired and checked by reading, not compiled, not run on hardware
//
#include <hip/hip_runtime.h>
#include <hip/hip_bf16.h>

// ---------------------------------------------------------------------------
// DyEdgeGAT on MI455X (gfx1250).
// Compute is dominated by two 100-step GRU recurrences (~77 GFLOP each of
// (M x 100) @ (100 x 300) matmul). Memory traffic (~100MB) is negligible vs
// 23.3 TB/s, so the win is: bf16 WMMA (v_wmma_f32_16x16x32_bf16) with the
// recurrent weight matrix resident in LDS (86KB of 320KB/WGP) across all
// time steps, gate math in fp32 out of an LDS strip.
// This revision: B-fragment LDS loads vectorized to ds_load_b128 and
// software-pipelined (double-buffered) so WMMA overlaps LDS latency instead
// of stalling on s_wait_dscnt 0x0 each tile.
// ---------------------------------------------------------------------------

#define B_SZ   256
#define NNODE  50
#define SEQ    100
#define HID    100
#define G3     300     // 3*HID
#define HP     128     // HID padded to K multiple of 32
#define CP     112     // per-gate chunk pitch (multiple of 16)
#define GP     336     // 3*CP, 21 N-tiles of 16
#define NT     (GP/16) // 21
#define TOPK   20
#define ROWS   64      // rows per block for WMMA kernels
#define MROWS  (B_SZ * NNODE)   // 12800

typedef __bf16 v16bf __attribute__((ext_vector_type(16)));
typedef float  v8f   __attribute__((ext_vector_type(8)));

union ABFrag { v16bf v; unsigned u[8]; uint4 q[2]; };

__device__ __forceinline__ unsigned short f2bf(float f) {
    unsigned u = __float_as_uint(f);
    unsigned r = u + 0x7FFFu + ((u >> 16) & 1u);   // round-to-nearest-even
    return (unsigned short)(r >> 16);
}

__device__ __forceinline__ float sigf(float x) { return 1.0f / (1.0f + __expf(-x)); }

// ---------------------------------------------------------------------------
// gh[64][336] = A[64][128](bf16, LDS) x B[336][128]^T (bf16, LDS, row g holds
// the K=128 reduction dim contiguously).  4 waves, wave w owns M-tile w.
// Fragment packing uses contiguous k = 16*(lane/16)+s for both A(16x32) and B(32x16).
// B fragments are 32 contiguous bytes -> 2x ds_load_b128, double-buffered.
// ---------------------------------------------------------------------------
__device__ __forceinline__ void mm_64x336x128(const unsigned short* __restrict__ abf,
                                              const unsigned short* __restrict__ wbf,
                                              float* __restrict__ gh,
                                              int lane, int wave) {
    const int m0 = wave * 16;
    const int mA = m0 + (lane & 15);
    const int kg = lane >> 4;          // lane group 0/1
    const int nl = lane & 15;          // B/D column within tile

    // Preload A fragments for the 4 K-tiles (reused for all 21 N-tiles).
    // Per lane: 32 contiguous bytes at k = kt*32 + kg*16.
    ABFrag a[4];
#pragma unroll
    for (int kt = 0; kt < 4; ++kt) {
        const uint4* pa = (const uint4*)(abf + mA * HP + kt * 32 + kg * 16);
        a[kt].q[0] = pa[0];
        a[kt].q[1] = pa[1];            // +16B (= +8 bf16): k += 8
    }

    // B fragment (j, kt): 32 contiguous bytes at row (j*16+nl), k = kt*32+kg*16.
    ABFrag bcur, bnxt;
    {
        const uint4* p = (const uint4*)(wbf + (size_t)nl * HP + kg * 16);
        bcur.q[0] = p[0];
        bcur.q[1] = p[1];
    }

    for (int j = 0; j < NT; ++j) {
        v8f acc = {};
#pragma unroll
        for (int kt = 0; kt < 4; ++kt) {
            // prefetch the next fragment while WMMA consumes the current one
            const int nj  = (kt == 3) ? (j + 1) : j;
            const int nkt = (kt + 1) & 3;
            if (nj < NT) {
                const uint4* p = (const uint4*)(wbf + (size_t)(nj * 16 + nl) * HP
                                                + nkt * 32 + kg * 16);
                bnxt.q[0] = p[0];
                bnxt.q[1] = p[1];
            }
            acc = __builtin_amdgcn_wmma_f32_16x16x32_bf16(
                false, a[kt].v, false, bcur.v, (short)0, acc, false, false);
            bcur = bnxt;               // register rename, no data movement
        }
        // D layout: vgpr i -> row m0+i (lanes 0-15) / m0+8+i (lanes 16-31)
#pragma unroll
        for (int i = 0; i < 8; ++i)
            gh[(m0 + i + kg * 8) * GP + j * 16 + nl] = acc[i];
    }
}

// ---------------------------------------------------------------------------
// K1: encoder GRU. grid = 50 nodes x 4 batch tiles, block = 128.
// ---------------------------------------------------------------------------
__global__ __launch_bounds__(128)
void encoder_gru(const float* __restrict__ x, const float* __restrict__ Wih,
                 const float* __restrict__ Whh, const float* __restrict__ bih,
                 const float* __restrict__ bhh, float* __restrict__ h_enc) {
    extern __shared__ char smem[];
    unsigned short* Wlds = (unsigned short*)smem;              // [GP][HP]
    unsigned short* hbf  = Wlds + GP * HP;                     // [64][HP]
    float* gh     = (float*)(hbf + ROWS * HP);                 // [64][GP]
    float* hstate = gh + ROWS * GP;                            // [64][CP]
    float* wih_l  = hstate + ROWS * CP;                        // [GP]
    float* bih_l  = wih_l + GP;
    float* bhh_l  = bih_l + GP;

    const int node  = blockIdx.x % NNODE;
    const int btile = blockIdx.x / NNODE;
    const int tid   = threadIdx.x;
    const int lane  = tid & 31, wave = tid >> 5;

    for (int i = tid; i < GP * HP; i += 128) Wlds[i] = 0;
    for (int i = tid; i < ROWS * HP; i += 128) hbf[i] = 0;
    for (int i = tid; i < ROWS * CP; i += 128) hstate[i] = 0;

    const float* Wn = Whh + (size_t)node * G3 * HID;
    __builtin_prefetch(Wn + tid, 0, 3);
    for (int i = tid; i < G3 * HID; i += 128) {
        int g = i / HID, h = i % HID;
        int c = g / HID, j = g % HID;
        Wlds[(c * CP + j) * HP + h] = f2bf(Wn[i]);
    }
    for (int i = tid; i < G3; i += 128) {
        int c = i / HID, j = i % HID, gp = c * CP + j;
        wih_l[gp] = Wih[(size_t)node * G3 + i];
        bih_l[gp] = bih[(size_t)node * G3 + i];
        bhh_l[gp] = bhh[(size_t)node * G3 + i];
    }
    __syncthreads();

    const int r0    = tid >> 1;             // row 0..63 for elementwise
    const int jbase = (tid & 1) * 50;
    const int brow  = btile * ROWS + r0;    // batch index
    const float* xrow = x + ((size_t)(brow * NNODE + node)) * SEQ;

    for (int t = 0; t < SEQ; ++t) {
        mm_64x336x128(hbf, Wlds, gh, lane, wave);
        __syncthreads();
        const float xv = xrow[t];
        for (int j = jbase; j < jbase + 50; ++j) {
            float gir = xv * wih_l[j]          + bih_l[j];
            float giz = xv * wih_l[CP + j]     + bih_l[CP + j];
            float gin = xv * wih_l[2 * CP + j] + bih_l[2 * CP + j];
            float ghr = gh[r0 * GP + j]          + bhh_l[j];
            float ghz = gh[r0 * GP + CP + j]     + bhh_l[CP + j];
            float ghn = gh[r0 * GP + 2 * CP + j] + bhh_l[2 * CP + j];
            float r = sigf(gir + ghr);
            float z = sigf(giz + ghz);
            float n = tanhf(gin + r * ghn);
            float hn = (1.0f - z) * n + z * hstate[r0 * CP + j];
            hstate[r0 * CP + j] = hn;
            hbf[r0 * HP + j] = f2bf(hn);
        }
        __syncthreads();
    }

    for (int i = tid; i < ROWS * HID; i += 128) {
        int r = i / HID, j = i % HID;
        h_enc[((size_t)((btile * ROWS + r) * NNODE + node)) * HID + j] = hstate[r * CP + j];
    }
}

// ---------------------------------------------------------------------------
// K2: batch-norm statistics over the batch axis (per (n,h)).
// ---------------------------------------------------------------------------
__global__ void bn_stats(const float* __restrict__ h_enc,
                         float* __restrict__ mean, float* __restrict__ var) {
    int idx = blockIdx.x * blockDim.x + threadIdx.x;
    if (idx >= NNODE * HID) return;
    int n = idx / HID, h = idx % HID;
    float s = 0.f, s2 = 0.f;
    for (int b = 0; b < B_SZ; ++b) {
        float v = h_enc[((size_t)(b * NNODE + n)) * HID + h];
        s += v; s2 += v * v;
    }
    float m = s * (1.0f / B_SZ);
    mean[idx] = m;
    var[idx]  = s2 * (1.0f / B_SZ) - m * m;
}

// K2b: apply BN + relu, emit bf16 rows padded to pitch 128 (zero pad).
__global__ void bn_apply(const float* __restrict__ h_enc,
                         const float* __restrict__ mean, const float* __restrict__ var,
                         const float* __restrict__ gamma, const float* __restrict__ beta,
                         unsigned short* __restrict__ hbnbf) {
    int idx = blockIdx.x * blockDim.x + threadIdx.x;
    if (idx >= MROWS * HP) return;
    int row = idx >> 7, k = idx & (HP - 1);
    float y = 0.f;
    if (k < HID) {
        int n = row % NNODE;
        int c = n * HID + k;
        float xv = h_enc[(size_t)row * HID + k];
        y = gamma[c] * (xv - mean[c]) * rsqrtf(var[c] + 1e-5f) + beta[c];
        y = fmaxf(y, 0.f);
    }
    hbnbf[idx] = f2bf(y);
}

// ---------------------------------------------------------------------------
// K3: xl = h@Wl+bl, xr = h@Wr+br, hp = h@Wg+bg as one WMMA pass
// (three 100x100 weights stacked into the 336-column B operand).
// ---------------------------------------------------------------------------
__global__ __launch_bounds__(128)
void gat_proj(const unsigned short* __restrict__ hbnbf,
              const float* __restrict__ Wl, const float* __restrict__ bl,
              const float* __restrict__ Wr, const float* __restrict__ br,
              const float* __restrict__ Wg, const float* __restrict__ bg,
              float* __restrict__ xl, float* __restrict__ xr, float* __restrict__ hp) {
    extern __shared__ char smem[];
    unsigned short* Wlds = (unsigned short*)smem;
    unsigned short* abf  = Wlds + GP * HP;
    float* gh = (float*)(abf + ROWS * HP);

    const int tid = threadIdx.x;
    const int lane = tid & 31, wave = tid >> 5;
    const int m0 = blockIdx.x * ROWS;

    for (int i = tid; i < GP * HP; i += 128) Wlds[i] = 0;
    for (int i = tid; i < 3 * HID * HID; i += 128) {
        int c = i / (HID * HID), r = i % (HID * HID);
        int h = r / HID, d = r % HID;
        const float* Wc = (c == 0) ? Wl : (c == 1) ? Wr : Wg;
        Wlds[(c * CP + d) * HP + h] = f2bf(Wc[h * HID + d]);   // B(k=h, n=d)
    }
    for (int i = tid; i < ROWS * HP; i += 128)
        abf[i] = hbnbf[(size_t)m0 * HP + i];
    __syncthreads();

    mm_64x336x128(abf, Wlds, gh, lane, wave);
    __syncthreads();

    for (int i = tid; i < ROWS * HID; i += 128) {
        int r = i / HID, d = i % HID;
        size_t o = (size_t)(m0 + r) * HID + d;
        xl[o] = gh[r * GP + d]          + bl[d];
        xr[o] = gh[r * GP + CP + d]     + br[d];
        hp[o] = gh[r * GP + 2 * CP + d] + bg[d];
    }
}

// ---------------------------------------------------------------------------
// K4: attention scores, softmax over 50 neighbors, zero diagonal, top-20.
// One block per (b,i).
// ---------------------------------------------------------------------------
__global__ void attention(const float* __restrict__ xl, const float* __restrict__ xr,
                          const float* __restrict__ att,
                          float* __restrict__ attnw, int* __restrict__ attni) {
    __shared__ float xli[HID];
    __shared__ float attL[HID];
    __shared__ float alpha[NNODE];
    const int b = blockIdx.x / NNODE;
    const int i = blockIdx.x % NNODE;
    const int tid = threadIdx.x;

    for (int d = tid; d < HID; d += 64) {
        xli[d]  = xl[(size_t)blockIdx.x * HID + d];
        attL[d] = att[d];
    }
    __syncthreads();
    if (tid < NNODE) {
        const float* xrj = xr + ((size_t)(b * NNODE + tid)) * HID;
        float a = 0.f;
        for (int d = 0; d < HID; ++d) {
            float e = xli[d] + xrj[d];
            e = (e > 0.f) ? e : 0.2f * e;     // leaky_relu(., 0.2)
            a += e * attL[d];
        }
        alpha[tid] = a;
    }
    __syncthreads();
    if (tid == 0) {
        float mx = -1e30f;
        for (int j = 0; j < NNODE; ++j) mx = fmaxf(mx, alpha[j]);
        float s = 0.f;
        for (int j = 0; j < NNODE; ++j) { alpha[j] = __expf(alpha[j] - mx); s += alpha[j]; }
        float inv = 1.0f / s;
        for (int j = 0; j < NNODE; ++j) alpha[j] *= inv;
        alpha[i] = 0.f;                        // * (1 - eye)
        for (int k = 0; k < TOPK; ++k) {       // iterative arg-max selection
            int jm = 0; float vm = alpha[0];
            for (int j = 1; j < NNODE; ++j)
                if (alpha[j] > vm) { vm = alpha[j]; jm = j; }
            attnw[(size_t)blockIdx.x * TOPK + k] = vm;
            attni[(size_t)blockIdx.x * TOPK + k] = jm;
            alpha[jm] = -1.f;
        }
    }
}

// ---------------------------------------------------------------------------
// K5: agg = relu(sum_k attn[k] * hp[b, idx[k], :]) -> bf16 rows (pitch 128).
// ---------------------------------------------------------------------------
__global__ void aggregate(const float* __restrict__ hp,
                          const float* __restrict__ attnw, const int* __restrict__ attni,
                          unsigned short* __restrict__ hnodebf) {
    __shared__ float w[TOPK];
    __shared__ int   id[TOPK];
    const int b = blockIdx.x / NNODE;
    const int tid = threadIdx.x;
    if (tid < TOPK) {
        w[tid]  = attnw[(size_t)blockIdx.x * TOPK + tid];
        id[tid] = attni[(size_t)blockIdx.x * TOPK + tid];
    }
    __syncthreads();
    float acc = 0.f;
    if (tid < HID) {
        for (int k = 0; k < TOPK; ++k)
            acc += w[k] * hp[((size_t)(b * NNODE + id[k])) * HID + tid];
        acc = fmaxf(acc, 0.f);
    }
    hnodebf[(size_t)blockIdx.x * HP + tid] = (tid < HID) ? f2bf(acc) : (unsigned short)0;
}

// ---------------------------------------------------------------------------
// K6: decoder gi = h_node @ dec_Wih^T + dec_bih  (WMMA), stored pitch 336.
// ---------------------------------------------------------------------------
__global__ __launch_bounds__(128)
void dec_gi_proj(const unsigned short* __restrict__ hnodebf,
                 const float* __restrict__ dWih, const float* __restrict__ dbih,
                 float* __restrict__ gi) {
    extern __shared__ char smem[];
    unsigned short* Wlds = (unsigned short*)smem;
    unsigned short* abf  = Wlds + GP * HP;
    float* gh = (float*)(abf + ROWS * HP);

    const int tid = threadIdx.x;
    const int lane = tid & 31, wave = tid >> 5;
    const int m0 = blockIdx.x * ROWS;

    for (int i = tid; i < GP * HP; i += 128) Wlds[i] = 0;
    __builtin_prefetch(dWih + tid, 0, 3);
    for (int i = tid; i < G3 * HID; i += 128) {
        int g = i / HID, h = i % HID;
        int c = g / HID, j = g % HID;
        Wlds[(c * CP + j) * HP + h] = f2bf(dWih[i]);           // B(k=h, n=g)
    }
    for (int i = tid; i < ROWS * HP; i += 128)
        abf[i] = hnodebf[(size_t)m0 * HP + i];
    __syncthreads();

    mm_64x336x128(abf, Wlds, gh, lane, wave);
    __syncthreads();

    for (int i = tid; i < ROWS * G3; i += 128) {
        int r = i / G3, jj = i % G3;
        int c = jj / HID, j = jj % HID;
        gi[(size_t)(m0 + r) * GP + c * CP + j] = gh[r * GP + c * CP + j] + dbih[jj];
    }
}

// ---------------------------------------------------------------------------
// K7: decoder GRU scan, 100 steps, fused relu(h)@fc_W + fc_b output.
// ---------------------------------------------------------------------------
__global__ __launch_bounds__(128)
void decoder_gru(const float* __restrict__ gi, const float* __restrict__ dWhh,
                 const float* __restrict__ dbhh, const float* __restrict__ fcW,
                 const float* __restrict__ fcb, float* __restrict__ out) {
    extern __shared__ char smem[];
    unsigned short* Wlds = (unsigned short*)smem;              // [GP][HP]
    unsigned short* hbf  = Wlds + GP * HP;                     // [64][HP]
    float* gh     = (float*)(hbf + ROWS * HP);                 // [64][GP]
    float* gil    = gh + ROWS * GP;                            // [64][GP]
    float* hstate = gil + ROWS * GP;                           // [64][CP]
    float* bhh_l  = hstate + ROWS * CP;                        // [GP]
    float* fc_l   = bhh_l + GP;                                // [CP]
    float* ypart  = fc_l + CP;                                 // [64][2]

    const int tid = threadIdx.x;
    const int lane = tid & 31, wave = tid >> 5;
    const int m0 = blockIdx.x * ROWS;

    for (int i = tid; i < GP * HP; i += 128) Wlds[i] = 0;
    for (int i = tid; i < ROWS * HP; i += 128) hbf[i] = 0;
    for (int i = tid; i < ROWS * CP; i += 128) hstate[i] = 0.f;
    __builtin_prefetch(dWhh + tid, 0, 3);
    for (int i = tid; i < G3 * HID; i += 128) {
        int g = i / HID, h = i % HID;
        int c = g / HID, j = g % HID;
        Wlds[(c * CP + j) * HP + h] = f2bf(dWhh[i]);
    }
    for (int i = tid; i < ROWS * GP; i += 128)
        gil[i] = gi[(size_t)m0 * GP + i];
    for (int i = tid; i < G3; i += 128) {
        int c = i / HID, j = i % HID;
        bhh_l[c * CP + j] = dbhh[i];
    }
    for (int i = tid; i < HID; i += 128) fc_l[i] = fcW[i];
    __syncthreads();

    const int r0    = tid >> 1;
    const int jbase = (tid & 1) * 50;
    const float fcb0 = fcb[0];

    for (int t = 0; t < SEQ; ++t) {
        mm_64x336x128(hbf, Wlds, gh, lane, wave);
        __syncthreads();
        float yp = 0.f;
        for (int j = jbase; j < jbase + 50; ++j) {
            float gir = gil[r0 * GP + j];
            float giz = gil[r0 * GP + CP + j];
            float gin = gil[r0 * GP + 2 * CP + j];
            float ghr = gh[r0 * GP + j]          + bhh_l[j];
            float ghz = gh[r0 * GP + CP + j]     + bhh_l[CP + j];
            float ghn = gh[r0 * GP + 2 * CP + j] + bhh_l[2 * CP + j];
            float r = sigf(gir + ghr);
            float z = sigf(giz + ghz);
            float n = tanhf(gin + r * ghn);
            float hn = (1.0f - z) * n + z * hstate[r0 * CP + j];
            hstate[r0 * CP + j] = hn;
            hbf[r0 * HP + j] = f2bf(hn);
            yp += fmaxf(hn, 0.f) * fc_l[j];
        }
        ypart[r0 * 2 + (tid & 1)] = yp;
        __syncthreads();
        if (tid < ROWS)
            out[(size_t)(m0 + tid) * SEQ + t] = ypart[tid * 2] + ypart[tid * 2 + 1] + fcb0;
        // no extra barrier needed: the post-mm barrier of the next iteration
        // orders this write before ypart is rewritten.
    }
}

// ---------------------------------------------------------------------------
// Host launcher
// ---------------------------------------------------------------------------
extern "C" void kernel_launch(void* const* d_in, const int* in_sizes, int n_in,
                              void* d_out, int out_size, void* d_ws, size_t ws_size,
                              hipStream_t stream) {
    const float* x     = (const float*)d_in[0];
    // d_in[1] = batch (int64), unused
    const float* gWih  = (const float*)d_in[2];
    const float* gWhh  = (const float*)d_in[3];
    const float* gbih  = (const float*)d_in[4];
    const float* gbhh  = (const float*)d_in[5];
    const float* gamma = (const float*)d_in[6];
    const float* beta  = (const float*)d_in[7];
    const float* Wl    = (const float*)d_in[8];
    const float* bl    = (const float*)d_in[9];
    const float* Wr    = (const float*)d_in[10];
    const float* br    = (const float*)d_in[11];
    const float* att   = (const float*)d_in[12];
    const float* Wg    = (const float*)d_in[13];
    const float* bg    = (const float*)d_in[14];
    const float* dWih  = (const float*)d_in[15];
    const float* dWhh  = (const float*)d_in[16];
    const float* dbih  = (const float*)d_in[17];
    const float* dbhh  = (const float*)d_in[18];
    const float* fcW   = (const float*)d_in[19];
    const float* fcb   = (const float*)d_in[20];
    float* out = (float*)d_out;

    // workspace carve (256B aligned)
    char* ws = (char*)d_ws;
    size_t off = 0;
    auto carve = [&](size_t bytes) -> void* {
        void* p = ws + off;
        off = (off + bytes + 255) & ~(size_t)255;
        return p;
    };
    float*          h_enc   = (float*)carve((size_t)MROWS * HID * 4);
    float*          bn_m    = (float*)carve((size_t)NNODE * HID * 4);
    float*          bn_v    = (float*)carve((size_t)NNODE * HID * 4);
    unsigned short* hbnbf   = (unsigned short*)carve((size_t)MROWS * HP * 2);
    float*          xl      = (float*)carve((size_t)MROWS * HID * 4);
    float*          xr      = (float*)carve((size_t)MROWS * HID * 4);
    float*          hp      = (float*)carve((size_t)MROWS * HID * 4);
    float*          attnw   = (float*)carve((size_t)MROWS * TOPK * 4);
    int*            attni   = (int*)carve((size_t)MROWS * TOPK * 4);
    unsigned short* hnodebf = (unsigned short*)carve((size_t)MROWS * HP * 2);
    float*          gi      = (float*)carve((size_t)MROWS * GP * 4);

    // dynamic LDS sizes
    const int ENC_LDS  = GP * HP * 2 + ROWS * HP * 2 + ROWS * GP * 4 + ROWS * CP * 4 + 3 * GP * 4;
    const int PROJ_LDS = GP * HP * 2 + ROWS * HP * 2 + ROWS * GP * 4;
    const int DEC_LDS  = GP * HP * 2 + ROWS * HP * 2 + 2 * ROWS * GP * 4 + ROWS * CP * 4
                       + GP * 4 + CP * 4 + ROWS * 2 * 4;

    (void)hipFuncSetAttribute((const void*)encoder_gru,
                              hipFuncAttributeMaxDynamicSharedMemorySize, ENC_LDS);
    (void)hipFuncSetAttribute((const void*)gat_proj,
                              hipFuncAttributeMaxDynamicSharedMemorySize, PROJ_LDS);
    (void)hipFuncSetAttribute((const void*)dec_gi_proj,
                              hipFuncAttributeMaxDynamicSharedMemorySize, PROJ_LDS);
    (void)hipFuncSetAttribute((const void*)decoder_gru,
                              hipFuncAttributeMaxDynamicSharedMemorySize, DEC_LDS);

    encoder_gru<<<NNODE * (B_SZ / ROWS), 128, ENC_LDS, stream>>>(x, gWih, gWhh, gbih, gbhh, h_enc);
    bn_stats<<<(NNODE * HID + 255) / 256, 256, 0, stream>>>(h_enc, bn_m, bn_v);
    bn_apply<<<(MROWS * HP + 255) / 256, 256, 0, stream>>>(h_enc, bn_m, bn_v, gamma, beta, hbnbf);
    gat_proj<<<MROWS / ROWS, 128, PROJ_LDS, stream>>>(hbnbf, Wl, bl, Wr, br, Wg, bg, xl, xr, hp);
    attention<<<MROWS, 64, 0, stream>>>(xl, xr, att, attnw, attni);
    aggregate<<<MROWS, 128, 0, stream>>>(hp, attnw, attni, hnodebf);
    dec_gi_proj<<<MROWS / ROWS, 128, PROJ_LDS, stream>>>(hnodebf, dWih, dbih, gi);
    decoder_gru<<<MROWS / ROWS, 128, DEC_LDS, stream>>>(gi, dWhh, dbhh, fcW, fcb, out);
}
